// GNN_42391327211901
// MI455X (gfx1250) — compile-verified, matrix operand order checked by reading, not run on hardware
//
#include <hip/hip_runtime.h>

// ---------------------------------------------------------------------------
// GraphSAGE 2-layer forward for MI455X (gfx1250, wave32, WMMA bf16)
//   N=100000 nodes, E=1600000 edges, D=128, H=128, O=64
//
// Roofline: edge gather/scatter dominates (~819MB reads + 205M f32 L2 atomics
// per layer; x and agg both fit in the 192MB L2). GEMMs are ~9.8 GFLOP total,
// executed as bf16 WMMA with f32 accumulation; A fragments are register-
// resident across all column tiles to avoid redundant global traffic.
// ---------------------------------------------------------------------------

typedef __attribute__((ext_vector_type(16))) __bf16 v16bf;
typedef __attribute__((ext_vector_type(8)))  float  v8f;

#define GNN_N 100000
#define GNN_E 1600000
#define GNN_D 128
#define GNN_H 128
#define GNN_O 64

// ---------------------------------------------------------------------------
// zero a float buffer (n4 = count of float4)
// ---------------------------------------------------------------------------
__global__ void gnn_zero_f32(float* __restrict__ p, long n4) {
    long i = (long)blockIdx.x * blockDim.x + threadIdx.x;
    if (i < n4) {
        ((float4*)p)[i] = make_float4(0.f, 0.f, 0.f, 0.f);
    }
}

// ---------------------------------------------------------------------------
// degree count: cnt[dst[e]] += 1
// ---------------------------------------------------------------------------
__global__ void gnn_count_deg(const int* __restrict__ dst, float* __restrict__ cnt, int E) {
    int e = blockIdx.x * blockDim.x + threadIdx.x;
    if (e < E) {
        unsafeAtomicAdd(&cnt[dst[e]], 1.0f);
    }
}

// ---------------------------------------------------------------------------
// scatter-add: agg[dst[e]][:] += X[src[e]][:]   (feature dim = 128)
// one edge per wave (32 lanes); edge id forced into an SGPR so the src/dst
// index loads are scalar; each lane owns a float4 chunk -> coalesced 512B row
// reads; f32 atomics resolve in L2 (51.2MB agg buffer << 192MB L2)
// ---------------------------------------------------------------------------
__global__ void gnn_scatter_add(const float* __restrict__ X,
                                const int* __restrict__ src,
                                const int* __restrict__ dst,
                                float* __restrict__ agg, int E) {
    // 256 threads = 8 waves = 8 edges per block
    int e = __builtin_amdgcn_readfirstlane(blockIdx.x * 8 + (threadIdx.x >> 5));
    if (e >= E) return;
    int c = (threadIdx.x & 31) * 4;
    int s = src[e];   // wave-uniform -> scalar load
    int d = dst[e];   // wave-uniform -> scalar load
    float4 v = *(const float4*)(X + (long)s * 128 + c);
    float* ap = agg + (long)d * 128 + c;
    unsafeAtomicAdd(ap + 0, v.x);
    unsafeAtomicAdd(ap + 1, v.y);
    unsafeAtomicAdd(ap + 2, v.z);
    unsafeAtomicAdd(ap + 3, v.w);
}

// ---------------------------------------------------------------------------
// pack f32 weights W[K=128][Nout] into wave32 bf16 B-fragments:
//   Bp[((ntile*4 + kstep)*32 + lane)][0..15]
// lane L (L&15 = column within tile, L>=16 selects the K 8..15 / 24..31 half)
// matches the ISA 16-bit 32x16 B striping -> GEMM reads 32B/lane/fragment.
// ---------------------------------------------------------------------------
__global__ void gnn_pack_w(const float* __restrict__ W, __bf16* __restrict__ Bp, int Nout) {
    int t = blockIdx.x * blockDim.x + threadIdx.x;
    int total = (Nout / 16) * 4 * 32;
    if (t >= total) return;
    int lane = t & 31;
    int ks   = (t >> 5) & 3;
    int nt   = t >> 7;
    int hi   = lane >> 4;
    int col  = nt * 16 + (lane & 15);
    int kb   = ks * 32 + hi * 8;
    __bf16* o = Bp + (long)t * 16;
#pragma unroll
    for (int j = 0; j < 8; ++j) o[j]     = (__bf16)W[(kb + j) * Nout + col];
#pragma unroll
    for (int j = 0; j < 8; ++j) o[8 + j] = (__bf16)W[(kb + 16 + j) * Nout + col];
}

// ---------------------------------------------------------------------------
// fused SAGE GEMM:  Y = act( (agg/deg) @ Wl + bias + X @ Wr )
// one wave owns a 16-row strip: it builds all 8 A fragments (4 agg-mean +
// 4 x, bf16, 64 VGPRs) ONCE, then loops over the NOUT/16 column tiles reusing
// them -> A global traffic is read-once. 8 WMMAs per column tile.
// Bias is folded into accumulator init.
// ---------------------------------------------------------------------------
template <int NOUT, bool RELU>
__global__ void gnn_sage_gemm(const float* __restrict__ Aagg,
                              const float* __restrict__ cnt,
                              const float* __restrict__ Ax,
                              const v16bf* __restrict__ Bl,   // packed
                              const v16bf* __restrict__ Br,   // packed
                              const float* __restrict__ bias,
                              float* __restrict__ Y) {
    constexpr int K  = 128;
    constexpr int NT = NOUT / 16;
    const int lane    = threadIdx.x & 31;
    const int wid     = threadIdx.x >> 5;           // 2 waves per block
    const int rowtile = blockIdx.x * 2 + wid;
    const int rowbase = rowtile * 16;
    const int hi      = lane >> 4;
    const int row     = rowbase + (lane & 15);      // A-fragment row for this lane
    const float inv   = 1.0f / fmaxf(cnt[row], 1.0f);

    const float* ag = Aagg + (long)row * K;
    const float* ax = Ax   + (long)row * K;

    // build A fragments once (register-resident across all column tiles)
    v16bf afrag[8];
#pragma unroll
    for (int ks = 0; ks < 4; ++ks) {
        const int k0 = ks * 32 + hi * 8;
        v16bf a;
#pragma unroll
        for (int j = 0; j < 8; ++j) a[j]     = (__bf16)(ag[k0 + j] * inv);
#pragma unroll
        for (int j = 0; j < 8; ++j) a[8 + j] = (__bf16)(ag[k0 + 16 + j] * inv);
        afrag[ks] = a;
    }
#pragma unroll
    for (int ks = 0; ks < 4; ++ks) {
        const int k0 = ks * 32 + hi * 8;
        v16bf a;
#pragma unroll
        for (int j = 0; j < 8; ++j) a[j]     = (__bf16)(ax[k0 + j]);
#pragma unroll
        for (int j = 0; j < 8; ++j) a[8 + j] = (__bf16)(ax[k0 + 16 + j]);
        afrag[4 + ks] = a;
    }

    // loop over column tiles, reusing A fragments
#pragma unroll
    for (int ct = 0; ct < NT; ++ct) {
        const int   col = ct * 16 + (lane & 15);
        const float bv  = bias[col];
        v8f acc;
#pragma unroll
        for (int r = 0; r < 8; ++r) acc[r] = bv;     // bias folded into init

#pragma unroll
        for (int ks = 0; ks < 4; ++ks) {
            v16bf b = Bl[(ct * 4 + ks) * 32 + lane];
            acc = __builtin_amdgcn_wmma_f32_16x16x32_bf16(false, afrag[ks], false, b,
                                                          (short)0, acc, false, false);
        }
#pragma unroll
        for (int ks = 0; ks < 4; ++ks) {
            v16bf b = Br[(ct * 4 + ks) * 32 + lane];
            acc = __builtin_amdgcn_wmma_f32_16x16x32_bf16(false, afrag[4 + ks], false, b,
                                                          (short)0, acc, false, false);
        }

#pragma unroll
        for (int r = 0; r < 8; ++r) {
            float v = acc[r];
            if (RELU) v = fmaxf(v, 0.0f);
            Y[(long)(rowbase + r + hi * 8) * NOUT + col] = v;
        }
    }
}

// ---------------------------------------------------------------------------
// launch
// ---------------------------------------------------------------------------
extern "C" void kernel_launch(void* const* d_in, const int* in_sizes, int n_in,
                              void* d_out, int out_size, void* d_ws, size_t ws_size,
                              hipStream_t stream) {
    (void)in_sizes; (void)n_in; (void)out_size; (void)ws_size;

    const float* x   = (const float*)d_in[0];
    const int*   ei  = (const int*)d_in[1];      // [2, E]: row0 = src, row1 = dst
    const float* W1l = (const float*)d_in[2];
    const float* b1  = (const float*)d_in[3];
    const float* W1r = (const float*)d_in[4];
    const float* W2l = (const float*)d_in[5];
    const float* b2  = (const float*)d_in[6];
    const float* W2r = (const float*)d_in[7];
    float*       out = (float*)d_out;

    const int* src = ei;
    const int* dst = ei + GNN_E;

    // workspace carve-up (all offsets 32B+ aligned)
    char*   ws    = (char*)d_ws;
    float*  cnt   = (float*)(ws + 0);                    // 400000 B
    float*  agg   = (float*)(ws + 524288);               // 51.2 MB (reused by layer 2)
    float*  h     = (float*)(ws + 51724288);             // 51.2 MB
    __bf16* W1l_p = (__bf16*)(ws + 102924288);           // 32 KB
    __bf16* W1r_p = (__bf16*)(ws + 102957056);           // 32 KB
    __bf16* W2l_p = (__bf16*)(ws + 102989824);           // 16 KB
    __bf16* W2r_p = (__bf16*)(ws + 103006208);           // 16 KB

    const long nFeat4 = (long)GNN_N * 128 / 4;
    const long nCnt4  = GNN_N / 4;

    // ---- weight packing (tiny) ----
    gnn_pack_w<<<(8 * 4 * 32 + 255) / 256, 256, 0, stream>>>(W1l, W1l_p, GNN_H);
    gnn_pack_w<<<(8 * 4 * 32 + 255) / 256, 256, 0, stream>>>(W1r, W1r_p, GNN_H);
    gnn_pack_w<<<(4 * 4 * 32 + 255) / 256, 256, 0, stream>>>(W2l, W2l_p, GNN_O);
    gnn_pack_w<<<(4 * 4 * 32 + 255) / 256, 256, 0, stream>>>(W2r, W2r_p, GNN_O);

    // ---- degrees ----
    gnn_zero_f32<<<(int)((nCnt4 + 255) / 256), 256, 0, stream>>>(cnt, nCnt4);
    gnn_count_deg<<<(GNN_E + 255) / 256, 256, 0, stream>>>(dst, cnt, GNN_E);

    // ---- layer 1: aggregate x, then h = relu(mean@W1l + b1 + x@W1r) ----
    gnn_zero_f32<<<(int)((nFeat4 + 255) / 256), 256, 0, stream>>>(agg, nFeat4);
    gnn_scatter_add<<<GNN_E / 8, 256, 0, stream>>>(x, src, dst, agg, GNN_E);
    {
        dim3 grid(GNN_N / 32);  // 2 waves (row tiles) per block
        gnn_sage_gemm<GNN_H, true><<<grid, 64, 0, stream>>>(
            agg, cnt, x, (const v16bf*)W1l_p, (const v16bf*)W1r_p, b1, h);
    }

    // ---- layer 2: aggregate h, then out = mean@W2l + b2 + h@W2r ----
    gnn_zero_f32<<<(int)((nFeat4 + 255) / 256), 256, 0, stream>>>(agg, nFeat4);
    gnn_scatter_add<<<GNN_E / 8, 256, 0, stream>>>(h, src, dst, agg, GNN_E);
    {
        dim3 grid(GNN_N / 32);
        gnn_sage_gemm<GNN_O, false><<<grid, 64, 0, stream>>>(
            agg, cnt, h, (const v16bf*)W2l_p, (const v16bf*)W2r_p, b2, out);
    }
}
